// RGAT_linear_22333829939346
// MI455X (gfx1250) — matrix operand order repaired and checked by reading, not compiled, not run on hardware
//
#include <hip/hip_runtime.h>

#define NN 100000
#define EE 400000
#define RR 3
#define INF_ 128
#define HIDF 128
#define NHEAD 4
#define DHEAD 32
#define OUTF 64

typedef __attribute__((ext_vector_type(16))) __bf16 bf16x16;
typedef __attribute__((ext_vector_type(8)))  float  f32x8;
typedef __attribute__((ext_vector_type(4)))  unsigned int u32x4;

union FragBF { bf16x16 v; u32x4 u[2]; };

__device__ __forceinline__ unsigned short f2bf(float f) {
    unsigned int u = __float_as_uint(f);
    unsigned int r = (u + 0x7FFFu + ((u >> 16) & 1u)) >> 16;
    return (unsigned short)r;
}

__device__ __forceinline__ void atomicMaxF(float* addr, float v) {
    // Ordered-int trick: works for mixed signs when initialized to -inf.
    if (v >= 0.0f) atomicMax((int*)addr, __float_as_int(v));
    else           atomicMin((unsigned int*)addr, __float_as_uint(v));
}

// ---------------- f32 -> bf16 conversion ----------------
__global__ void k_f32_to_bf16(const float* __restrict__ in,
                              unsigned short* __restrict__ outp, int total) {
    int t = blockIdx.x * blockDim.x + threadIdx.x;
    if (t >= total) return;
    outp[t] = f2bf(in[t]);
}

// ---------------- fill float ----------------
__global__ void k_fill_f32(float* __restrict__ p, float val, int total) {
    int t = blockIdx.x * blockDim.x + threadIdx.x;
    if (t >= total) return;
    p[t] = val;
}

// ---------------- pack W (f32 row-major [K, NCOLS]) into WMMA B fragments ----
// B fragment layout used here (contiguous k, same order as the A loads in k_gemm_bf16) (32xK -> 16 cols tile):
//   lanes 0-15:  N = n0 + lane,    halfs i=0..15 -> K = k0 + i
//   lanes 16-31: N = n0 + lane-16, halfs i=0..15 -> K = k0 + 16 + i
__global__ void k_pack_w(const float* __restrict__ W, unsigned short* __restrict__ outp,
                         int ncols, int NT, int total) {
    int t = blockIdx.x * blockDim.x + threadIdx.x;
    if (t >= total) return;
    int lane = t & 31;
    int tile = t >> 5;
    int kt = tile / NT;
    int nt = tile - kt * NT;
    int hi = lane >> 4;
    int col = nt * 16 + (lane & 15);
    int kbase = kt * 32 + hi * 16;
    unsigned short* op = outp + ((size_t)tile * 32 + lane) * 16;
#pragma unroll
    for (int i = 0; i < 16; ++i)
        op[i] = f2bf(W[(size_t)(kbase + i) * ncols + col]);
}

// ---------------- GEMM: C[M,NCOLS] = A[M,K](bf16) * Bpacked + bias ---------
// One 16x16 output tile per wave; blockDim.x = (NCOLS/16)*32.
__global__ void k_gemm_bf16(const unsigned short* __restrict__ A,
                            const unsigned short* __restrict__ Bp,
                            float* __restrict__ C,
                            const float* __restrict__ bias,
                            int K, int ncols) {
    const int lane = threadIdx.x & 31;
    const int wave = threadIdx.x >> 5;          // = column-tile index nt
    const int NT   = ncols >> 4;
    const int row0 = blockIdx.x * 16;
    const int hi   = lane >> 4;
    const int l15  = lane & 15;
    const int arow = row0 + l15;

    f32x8 c = {};
    const int KT = K >> 5;
    for (int kt = 0; kt < KT; ++kt) {
        FragBF a, b;
        // A fragment: lanes 0-15 take K = k0+0..15; lanes 16-31: K = k0+16..31
        const unsigned short* ap = A + (size_t)arow * K + kt * 32 + hi * 16;
        a.u[0] = *(const u32x4*)(ap);
        a.u[1] = *(const u32x4*)(ap + 8);
        const unsigned short* bp = Bp + (((size_t)kt * NT + wave) * 32 + lane) * 16;
        b.u[0] = *(const u32x4*)(bp);
        b.u[1] = *(const u32x4*)(bp + 8);
        c = __builtin_amdgcn_wmma_f32_16x16x32_bf16(
                false, a.v, false, b.v, (short)0, c, false, false);
    }
    const int col = wave * 16 + l15;
    const float bv = bias ? bias[col] : 0.0f;
#pragma unroll
    for (int v = 0; v < 8; ++v) {
        int m = row0 + v + hi * 8;   // C layout: VGPR v -> M=v (lanes<16), M=v+8 (lanes>=16)
        C[(size_t)m * ncols + col] = c[v] + bv;
    }
}

// ---------------- per-node attention logits ----------------
__global__ void k_el_er(const float* __restrict__ z,
                        const float* __restrict__ al, const float* __restrict__ ar,
                        float* __restrict__ el, float* __restrict__ er, int total) {
    int t = blockIdx.x * blockDim.x + threadIdx.x;
    if (t >= total) return;
    int n = t >> 2, h = t & 3;
    const float* zp  = z + (size_t)n * HIDF + h * DHEAD;
    const float* alp = al + h * DHEAD;
    const float* arp = ar + h * DHEAD;
    float sl = 0.f, sr = 0.f;
#pragma unroll
    for (int d = 0; d < DHEAD; ++d) { float zv = zp[d]; sl += zv * alp[d]; sr += zv * arp[d]; }
    el[t] = sl; er[t] = sr;
}

// ---------------- edge pass A: segment max ----------------
__global__ void k_edge_max(const int* __restrict__ src, const int* __restrict__ dst,
                           const float* __restrict__ el, const float* __restrict__ er,
                           float* __restrict__ m, int total) {
    int t = blockIdx.x * blockDim.x + threadIdx.x;
    if (t >= total) return;
    int e = t >> 2, h = t & 3;
    int s = src[e], d = dst[e];
    float v = el[s * 4 + h] + er[d * 4 + h];
    v = v > 0.f ? v : 0.2f * v;            // GAT leaky relu
    atomicMaxF(&m[d * 4 + h], v);
}

// ---------------- edge pass B: exp + weighted scatter (wave per edge) -------
__global__ void k_edge_scatter(const int* __restrict__ src, const int* __restrict__ dst,
                               const float* __restrict__ el, const float* __restrict__ er,
                               const float* __restrict__ m, const float* __restrict__ z,
                               float* __restrict__ ssum, float* __restrict__ acc) {
    int gt = blockIdx.x * blockDim.x + threadIdx.x;
    int e = gt >> 5;
    int lane = gt & 31;
    if (e >= EE) return;
    int s = src[e], d = dst[e];
    float ex0, ex1, ex2, ex3;
    {
        float v0 = el[s * 4 + 0] + er[d * 4 + 0]; v0 = v0 > 0.f ? v0 : 0.2f * v0;
        float v1 = el[s * 4 + 1] + er[d * 4 + 1]; v1 = v1 > 0.f ? v1 : 0.2f * v1;
        float v2 = el[s * 4 + 2] + er[d * 4 + 2]; v2 = v2 > 0.f ? v2 : 0.2f * v2;
        float v3 = el[s * 4 + 3] + er[d * 4 + 3]; v3 = v3 > 0.f ? v3 : 0.2f * v3;
        ex0 = __expf(v0 - m[d * 4 + 0]);
        ex1 = __expf(v1 - m[d * 4 + 1]);
        ex2 = __expf(v2 - m[d * 4 + 2]);
        ex3 = __expf(v3 - m[d * 4 + 3]);
    }
    if (lane < 4) {
        float exv = (lane == 0) ? ex0 : (lane == 1) ? ex1 : (lane == 2) ? ex2 : ex3;
        atomicAdd(&ssum[d * 4 + lane], exv);
    }
    const float* zs = z + (size_t)s * HIDF;
    float* ad = acc + (size_t)d * HIDF;
    atomicAdd(&ad[lane],      ex0 * zs[lane]);
    atomicAdd(&ad[32 + lane], ex1 * zs[32 + lane]);
    atomicAdd(&ad[64 + lane], ex2 * zs[64 + lane]);
    atomicAdd(&ad[96 + lane], ex3 * zs[96 + lane]);
}

// ---------------- finalize per relation: hacc += acc/ssum + bias ------------
__global__ void k_finalize(const float* __restrict__ acc, const float* __restrict__ ssum,
                           const float* __restrict__ bias, float* __restrict__ hacc,
                           int total) {
    int t = blockIdx.x * blockDim.x + threadIdx.x;
    if (t >= total) return;
    int n = t >> 7;
    int f = t & 127;
    int h = f >> 5;
    float sv = ssum[n * 4 + h];
    float v = sv > 0.f ? acc[t] / sv : 0.f;   // isolated nodes contribute 0 (matches segment_sum)
    hacc[t] += v + bias[f];
}

// ---------------- mean over relations + activation -> next-layer bf16 -------
__global__ void k_act_bf16(const float* __restrict__ hacc, unsigned short* __restrict__ hb,
                           int total, int apply_act) {
    int t = blockIdx.x * blockDim.x + threadIdx.x;
    if (t >= total) return;
    float v = hacc[t] * (1.0f / (float)RR);
    if (apply_act) v = v > 0.f ? v : 0.01f * v;
    hb[t] = f2bf(v);
}

extern "C" void kernel_launch(void* const* d_in, const int* in_sizes, int n_in,
                              void* d_out, int out_size, void* d_ws, size_t ws_size,
                              hipStream_t stream) {
    const float* x    = (const float*)d_in[0];
    const int*   src  = (const int*)d_in[1];
    const int*   dst  = (const int*)d_in[2];
    const float* W0   = (const float*)d_in[3];
    const float* al0  = (const float*)d_in[4];
    const float* ar0  = (const float*)d_in[5];
    const float* b0   = (const float*)d_in[6];
    const float* W12  = (const float*)d_in[7];
    const float* al12 = (const float*)d_in[8];
    const float* ar12 = (const float*)d_in[9];
    const float* b12  = (const float*)d_in[10];
    const float* Wlin = (const float*)d_in[11];
    const float* blin = (const float*)d_in[12];
    float* outp = (float*)d_out;

    // -------- workspace carve --------
    char* ws = (char*)d_ws;
    size_t off = 0;
    auto carve = [&](size_t bytes) -> char* {
        char* p = ws + off; off += (bytes + 255) & ~(size_t)255; return p;
    };
    unsigned short* hbf   = (unsigned short*)carve((size_t)NN * HIDF * 2);
    float* zbuf           = (float*)carve((size_t)NN * HIDF * 4);
    float* acc            = (float*)carve((size_t)NN * HIDF * 4);
    float* hacc           = (float*)carve((size_t)NN * HIDF * 4);
    float* el             = (float*)carve((size_t)NN * NHEAD * 4);
    float* er             = (float*)carve((size_t)NN * NHEAD * 4);
    float* mbuf           = (float*)carve((size_t)NN * NHEAD * 4);
    float* ssum           = (float*)carve((size_t)NN * NHEAD * 4);
    unsigned short* wpack = (unsigned short*)carve((size_t)RR * 128 * 128 * 2);
    unsigned short* wlinp = (unsigned short*)carve((size_t)128 * 64 * 2);

    const int TPB = 256;
    const int totNF = NN * HIDF;          // 12.8M
    const int totNH = NN * NHEAD;         // 400k
    const int totEH = EE * NHEAD;         // 1.6M
    const float NEG_INF = __builtin_huge_valf() * -1.0f;

    // x -> bf16 activations
    k_f32_to_bf16<<<(totNF + TPB - 1) / TPB, TPB, 0, stream>>>(x, hbf, totNF);

    for (int layer = 0; layer < 3; ++layer) {
        const float* Wl = (layer == 0) ? W0  : W12  + (size_t)(layer - 1) * RR * 128 * 128;
        const float* aL = (layer == 0) ? al0 : al12 + (size_t)(layer - 1) * RR * 128;
        const float* aR = (layer == 0) ? ar0 : ar12 + (size_t)(layer - 1) * RR * 128;
        const float* bL = (layer == 0) ? b0  : b12  + (size_t)(layer - 1) * RR * 128;

        // pack relation weights into WMMA B-fragment layout (bf16)
        for (int r = 0; r < RR; ++r) {
            int tw = (128 / 32) * (128 / 16) * 32;   // KT*NT*32 = 1024 threads
            k_pack_w<<<(tw + TPB - 1) / TPB, TPB, 0, stream>>>(
                Wl + (size_t)r * 128 * 128, wpack + (size_t)r * 128 * 128, 128, 8, tw);
        }
        hipMemsetAsync(hacc, 0, (size_t)NN * HIDF * 4, stream);

        for (int r = 0; r < RR; ++r) {
            // z = h @ W_r   (WMMA bf16 -> f32)
            k_gemm_bf16<<<NN / 16, (HIDF / 16) * 32, 0, stream>>>(
                hbf, wpack + (size_t)r * 128 * 128, zbuf, nullptr, HIDF, HIDF);
            // attention logits
            k_el_er<<<(totNH + TPB - 1) / TPB, TPB, 0, stream>>>(
                zbuf, aL + r * 128, aR + r * 128, el, er, totNH);
            // softmax state init
            k_fill_f32<<<(totNH + TPB - 1) / TPB, TPB, 0, stream>>>(mbuf, NEG_INF, totNH);
            hipMemsetAsync(ssum, 0, (size_t)NN * NHEAD * 4, stream);
            hipMemsetAsync(acc, 0, (size_t)NN * HIDF * 4, stream);
            // segment max
            k_edge_max<<<(totEH + TPB - 1) / TPB, TPB, 0, stream>>>(
                src + (size_t)r * EE, dst + (size_t)r * EE, el, er, mbuf, totEH);
            // exp + weighted scatter-sum (wave per edge)
            k_edge_scatter<<<(EE * 32) / TPB, TPB, 0, stream>>>(
                src + (size_t)r * EE, dst + (size_t)r * EE, el, er, mbuf, zbuf, ssum, acc);
            // normalize + bias, accumulate into hetero sum
            k_finalize<<<(totNF + TPB - 1) / TPB, TPB, 0, stream>>>(
                acc, ssum, bL + r * 128, hacc, totNF);
        }
        // mean over relations (+ leaky_relu for layers 0,1) -> bf16 for next GEMM
        k_act_bf16<<<(totNF + TPB - 1) / TPB, TPB, 0, stream>>>(
            hacc, hbf, totNF, (layer < 2) ? 1 : 0);
    }

    // final linear: out = h @ Wlin + blin   [N,128] x [128,64]
    {
        int tw = (128 / 32) * (64 / 16) * 32;    // 512 threads
        k_pack_w<<<(tw + TPB - 1) / TPB, TPB, 0, stream>>>(Wlin, wlinp, 64, 4, tw);
        k_gemm_bf16<<<NN / 16, (OUTF / 16) * 32, 0, stream>>>(
            hbf, wlinp, outp, blin, HIDF, OUTF);
    }
}
